// cknrm_39917426049515
// MI455X (gfx1250) — matrix operand order repaired and checked by reading, not compiled, not run on hardware
//
#include <hip/hip_runtime.h>

typedef __attribute__((ext_vector_type(16))) _Float16 v16h;
typedef __attribute__((ext_vector_type(8)))  _Float16 v8h;
typedef __attribute__((ext_vector_type(8)))  float    v8f;

union AFrag { v16h v; v8h h[2]; };

#define WMMA_F16(a, b, c) \
  __builtin_amdgcn_wmma_f32_16x16x32_f16(false, (a), false, (b), (short)0, (c), false, false)

// ---------------------------------------------------------------------------
// Stage f32 embeddings -> f16, channel dim padded 300 -> 320 (zeros), plus
// 2 extra zero rows at the tail so bi/tri sliding windows never read OOB.
// ---------------------------------------------------------------------------
__global__ void cvt_embed_kernel(const float* __restrict__ src,
                                 _Float16* __restrict__ dst, int nrows) {
  const int row = blockIdx.x;       // grid = nrows + 2
  const int c   = threadIdx.x;      // block = 320
  float v = 0.0f;
  if (row < nrows && c < 300) v = src[(long)row * 300 + c];
  dst[(long)row * 320 + c] = (_Float16)v;
}

// Weights (128, taps, 300) -> f16 (128, taps*320), each tap zero-padded to 320.
__global__ void cvt_weight_kernel(const float* __restrict__ w,
                                  _Float16* __restrict__ dst, int taps) {
  const int n = blockIdx.x;         // grid = 128
  const int c = threadIdx.x;        // block = 320
  for (int t = 0; t < taps; ++t) {
    float v = (c < 300) ? w[((long)n * taps + t) * 300 + c] : 0.0f;
    dst[(long)n * taps * 320 + (long)t * 320 + c] = (_Float16)v;
  }
}

// ---------------------------------------------------------------------------
// Fused n-gram conv (as GEMM via WMMA) + bias + ReLU + L2-normalize -> f16.
// One wave = one 16-row x 128-channel output block. K = Ktiles*32.
// Weights are staged once per workgroup into LDS with CDNA5 async
// global->LDS copies (ASYNCcnt), then B-fragments are read from LDS.
// ---------------------------------------------------------------------------
__launch_bounds__(256)
__global__ void conv_norm_kernel(const _Float16* __restrict__ E,
                                 const _Float16* __restrict__ W,
                                 const float* __restrict__ bias,
                                 _Float16* __restrict__ Out,
                                 int Ktiles) {
  extern __shared__ _Float16 Wlds[];               // Kg*128 halfs
  const int Kg = Ktiles * 32;

  // ---- async-stage the whole weight matrix into LDS (16B chunks/lane) ----
  {
    const int nchunks = (Kg * 128) / 8;            // 8 halfs = 16 bytes each
    for (int i = threadIdx.x; i < nchunks; i += 256) {
      const _Float16* gp = W + (long)i * 8;
      unsigned lp = (unsigned)(size_t)(Wlds + (long)i * 8);  // LDS byte offset
      asm volatile("global_load_async_to_lds_b128 %0, %1, off"
                   :: "v"(lp), "v"(gp) : "memory");
    }
    asm volatile("s_wait_asynccnt 0x0" ::: "memory");
  }
  __syncthreads();

  const int lane  = threadIdx.x & 31;
  const int wave  = threadIdx.x >> 5;
  const int l15   = lane & 15;
  const int hi    = lane >> 4;                     // 0: rows 0-7 | 1: rows 8-15
  const long mtile = (long)blockIdx.x * 8 + wave;
  const long rowbase = mtile * 16;

  v8f acc[8] = {};                                 // 8 N-tiles of 16 channels

  const _Float16* arow = E + (rowbase + l15) * 320;
  for (int kt = 0; kt < Ktiles; ++kt) {
    AFrag a;
    const _Float16* ap = arow + kt * 32 + hi * 16; // A 16x32, contiguous k = 16h + s (matches B)
    a.h[0] = *(const v8h*)(ap);
    a.h[1] = *(const v8h*)(ap + 8);
#pragma unroll
    for (int nt = 0; nt < 8; ++nt) {
      const _Float16* bp = Wlds + (long)(nt * 16 + l15) * Kg + kt * 32 + hi * 16;
      v16h bf = *(const v16h*)bp;                  // B 32x16, contiguous k = 16h + s (same as A)
      acc[nt] = WMMA_F16(a.v, bf, acc[nt]);
    }
  }

  // bias + relu + 1e-9, per-row sum of squares
  float ss[8];
#pragma unroll
  for (int r = 0; r < 8; ++r) ss[r] = 0.0f;
#pragma unroll
  for (int nt = 0; nt < 8; ++nt) {
    const float bv = bias[nt * 16 + l15];
#pragma unroll
    for (int r = 0; r < 8; ++r) {
      float v = acc[nt][r] + bv;
      v = fmaxf(v, 0.0f) + 1e-9f;
      acc[nt][r] = v;
      ss[r] += v * v;
    }
  }
  // reduce over the 16 column-lanes of each half-wave
#pragma unroll
  for (int r = 0; r < 8; ++r) {
    float s = ss[r];
    s += __shfl_xor(s, 1);
    s += __shfl_xor(s, 2);
    s += __shfl_xor(s, 4);
    s += __shfl_xor(s, 8);
    ss[r] = 1.0f / fmaxf(sqrtf(s), 1e-10f);
  }
  const long rowb = rowbase + hi * 8;
#pragma unroll
  for (int r = 0; r < 8; ++r) {
    _Float16* op = Out + (rowb + r) * 128 + l15;
#pragma unroll
    for (int nt = 0; nt < 8; ++nt)
      op[nt * 16] = (_Float16)(acc[nt][r] * ss[r]);
  }
}

// ---------------------------------------------------------------------------
// One block per (batch, combo): sim GEMM (K=128, 4 WMMA per 16x16 tile) fused
// with 11 RBF kernels, masked doc pooling, log, masked query sum -> 11 feats.
// ---------------------------------------------------------------------------
__launch_bounds__(256)
__global__ void intersect_kernel(const _Float16* __restrict__ Qn0,
                                 const _Float16* __restrict__ Qn1,
                                 const _Float16* __restrict__ Qn2,
                                 const _Float16* __restrict__ Dn0,
                                 const _Float16* __restrict__ Dn1,
                                 const _Float16* __restrict__ Dn2,
                                 const float* __restrict__ qwm,
                                 const float* __restrict__ dwm,
                                 float* __restrict__ feats) {
  __shared__ float pool[8][32][11];
  __shared__ float logs[352];

  const int b = blockIdx.x;
  const int c = blockIdx.y;
  const int qselA[9] = {0, 0, 0, 1, 2, 1, 1, 2, 2};
  const int dselA[9] = {0, 2, 1, 0, 0, 1, 2, 1, 2};
  const int qg = qselA[c], dg = dselA[c];
  const _Float16* Qs[3] = {Qn0, Qn1, Qn2};
  const _Float16* Ds[3] = {Dn0, Dn1, Dn2};
  const _Float16* Q = Qs[qg] + (long)b * 32 * 128;
  const _Float16* D = Ds[dg] + (long)b * 1024 * 128;
  const int LQg = 32 - qg, LDg = 1024 - dg;

  const float MUa[11]  = {1.0f, 0.9f, 0.7f, 0.5f, 0.3f, 0.1f,
                          -0.1f, -0.3f, -0.5f, -0.7f, -0.9f};
  const float NEGa[11] = {-500000.0f, -50.0f, -50.0f, -50.0f, -50.0f, -50.0f,
                          -50.0f, -50.0f, -50.0f, -50.0f, -50.0f};

  for (int i = threadIdx.x; i < 8 * 32 * 11; i += 256) (&pool[0][0][0])[i] = 0.0f;
  __syncthreads();

  const int lane = threadIdx.x & 31;
  const int wave = threadIdx.x >> 5;
  const int l15  = lane & 15;
  const int hi   = lane >> 4;
  const int mtile = wave & 1;                // query-row tile: rows 0-15 / 16-31

  // preload the 4 A fragments (rows fixed for this wave)
  AFrag a[4];
  const _Float16* qrow = Q + (long)(mtile * 16 + l15) * 128;
#pragma unroll
  for (int kt = 0; kt < 4; ++kt) {
    const _Float16* ap = qrow + kt * 32 + hi * 16;
    a[kt].h[0] = *(const v8h*)ap;
    a[kt].h[1] = *(const v8h*)(ap + 8);
  }

  float acc[8][11] = {};
  for (int nt = (wave >> 1); nt < 64; nt += 4) {   // 64 doc tiles of 16
    const int p = nt * 16 + l15;                   // doc position of this lane's column
    const float md = (p < LDg) ? dwm[b * 1024 + p] : 0.0f;
    v8f cf = {};
    const _Float16* drow = D + (long)p * 128;
#pragma unroll
    for (int kt = 0; kt < 4; ++kt) {
      v16h bf = *(const v16h*)(drow + kt * 32 + hi * 16);
      cf = WMMA_F16(a[kt].v, bf, cf);
    }
    if (md != 0.0f) {
#pragma unroll
      for (int r = 0; r < 8; ++r) {
        const float s = cf[r];
#pragma unroll
        for (int k = 0; k < 11; ++k) {
          const float d2 = s - MUa[k];
          acc[r][k] += md * __expf(d2 * d2 * NEGa[k]);
        }
      }
    }
  }

  // reduce over the 16 column-lanes of each half-wave
#pragma unroll
  for (int r = 0; r < 8; ++r) {
#pragma unroll
    for (int k = 0; k < 11; ++k) {
      float v = acc[r][k];
      v += __shfl_xor(v, 1);
      v += __shfl_xor(v, 2);
      v += __shfl_xor(v, 4);
      v += __shfl_xor(v, 8);
      acc[r][k] = v;
    }
  }
  if (l15 == 0) {
#pragma unroll
    for (int r = 0; r < 8; ++r)
#pragma unroll
      for (int k = 0; k < 11; ++k)
        pool[wave][mtile * 16 + hi * 8 + r][k] = acc[r][k];
  }
  __syncthreads();

  for (int t = threadIdx.x; t < 352; t += 256) {
    const int m = t / 11, k = t % 11;
    float s = 0.0f;
#pragma unroll
    for (int w = 0; w < 8; ++w) s += pool[w][m][k];
    const float qm = (m < LQg) ? qwm[b * 32 + m] : 0.0f;
    logs[t] = __logf(fmaxf(s, 1e-10f)) * qm;
  }
  __syncthreads();

  if (threadIdx.x < 11) {
    float f = 0.0f;
    for (int m = 0; m < 32; ++m) f += logs[m * 11 + threadIdx.x];
    feats[(long)b * 99 + c * 11 + threadIdx.x] = f;
  }
}

// score[b] = feats[b,:] @ w_dense + b_dense
__global__ void dense_kernel(const float* __restrict__ feats,
                             const float* __restrict__ wd,
                             const float* __restrict__ bd,
                             float* __restrict__ score) {
  const int b = threadIdx.x;
  if (b < 64) {
    float s = bd[0];
    for (int j = 0; j < 99; ++j) s += feats[b * 99 + j] * wd[j];
    score[b] = s;
  }
}

extern "C" void kernel_launch(void* const* d_in, const int* in_sizes, int n_in,
                              void* d_out, int out_size, void* d_ws, size_t ws_size,
                              hipStream_t stream) {
  (void)in_sizes; (void)n_in; (void)out_size; (void)ws_size;
  const float* qw      = (const float*)d_in[0];
  const float* dw      = (const float*)d_in[1];
  const float* qwm     = (const float*)d_in[2];
  const float* dwm     = (const float*)d_in[3];
  const float* w_uni   = (const float*)d_in[4];
  const float* b_uni   = (const float*)d_in[5];
  const float* w_bi    = (const float*)d_in[6];
  const float* b_bi    = (const float*)d_in[7];
  const float* w_tri   = (const float*)d_in[8];
  const float* b_tri   = (const float*)d_in[9];
  const float* w_dense = (const float*)d_in[10];
  const float* b_dense = (const float*)d_in[11];

  const int MQ = 64 * 32;      // 2048 query rows
  const int MD = 64 * 1024;    // 65536 doc rows

  char* base = (char*)d_ws;
  size_t off = 0;
  auto take = [&](size_t bytes) -> char* {
    char* p = base + off;
    off = (off + bytes + 255) & ~(size_t)255;
    return p;
  };
  _Float16* Eq = (_Float16*)take((size_t)(MQ + 2) * 320 * 2);
  _Float16* Ed = (_Float16*)take((size_t)(MD + 2) * 320 * 2);
  _Float16* Wn[3];
  Wn[0] = (_Float16*)take((size_t)128 * 320 * 2);
  Wn[1] = (_Float16*)take((size_t)128 * 640 * 2);
  Wn[2] = (_Float16*)take((size_t)128 * 960 * 2);
  _Float16* Qn[3];
  _Float16* Dn[3];
  for (int g = 0; g < 3; ++g) Qn[g] = (_Float16*)take((size_t)MQ * 128 * 2);
  for (int g = 0; g < 3; ++g) Dn[g] = (_Float16*)take((size_t)MD * 128 * 2);

  cvt_embed_kernel<<<MQ + 2, 320, 0, stream>>>(qw, Eq, MQ);
  cvt_embed_kernel<<<MD + 2, 320, 0, stream>>>(dw, Ed, MD);
  cvt_weight_kernel<<<128, 320, 0, stream>>>(w_uni, Wn[0], 1);
  cvt_weight_kernel<<<128, 320, 0, stream>>>(w_bi,  Wn[1], 2);
  cvt_weight_kernel<<<128, 320, 0, stream>>>(w_tri, Wn[2], 3);

  const float* biases[3] = {b_uni, b_bi, b_tri};
  for (int g = 0; g < 3; ++g) {
    const size_t shbytes = (size_t)(10 * (g + 1)) * 32 * 128 * 2;  // Kg*128 halfs
    conv_norm_kernel<<<MQ / 128, 256, shbytes, stream>>>(Eq, Wn[g], biases[g], Qn[g], 10 * (g + 1));
    conv_norm_kernel<<<MD / 128, 256, shbytes, stream>>>(Ed, Wn[g], biases[g], Dn[g], 10 * (g + 1));
  }

  float* score = (float*)d_out;
  float* feats = (float*)d_out + 64;
  intersect_kernel<<<dim3(64, 9), 256, 0, stream>>>(Qn[0], Qn[1], Qn[2],
                                                    Dn[0], Dn[1], Dn[2],
                                                    qwm, dwm, feats);
  dense_kernel<<<1, 64, 0, stream>>>(feats, w_dense, b_dense, score);
}
